// LorentzMLA_68289980007157
// MI455X (gfx1250) — matrix operand order repaired and checked by reading, not compiled, not run on hardware
//
#include <hip/hip_runtime.h>
#include <hip/hip_bf16.h>
#include <cstdint>
#include <cstddef>

// ---------------------------------------------------------------------------
// LorentzMLA for MI455X (gfx1250): bf16 WMMA GEMMs + flash attention.
// All matrix math on v_wmma_f32_16x16x32_bf16; epilogues in f32 VALU.
// GEMM wave tile = 32(M) x 64(N): each B fragment feeds 2 WMMAs, each A
// fragment feeds 4 -> ~21 FLOP/byte from L2 (entire working set is
// L2-resident: ~120MB hot vs 192MB L2).
// ---------------------------------------------------------------------------

typedef unsigned short u16;
typedef __attribute__((ext_vector_type(16))) __bf16 bf16x16;
typedef __attribute__((ext_vector_type(8)))  float   v8f;
typedef __attribute__((ext_vector_type(8)))  u16     us8;
typedef __attribute__((ext_vector_type(16))) u16     us16;

#define SCALE_QK 0.07198158f   /* (192+1)^-0.5 */

__device__ __forceinline__ u16 f2bf(float f) {
  unsigned u = __builtin_bit_cast(unsigned, f);
  unsigned r = u + 0x7fffu + ((u >> 16) & 1u);   // round-nearest-even
  return (u16)(r >> 16);
}

// A-fragment (16x32 bf16, MxK): lane r=l&15 holds row M=r; half=l>>4 selects
// K range [half*16, half*16+15] (contiguous, matching load_bfrag)
__device__ __forceinline__ bf16x16 load_afrag(const u16* rowk0, int half) {
  us8 c0 = *reinterpret_cast<const us8*>(rowk0 + half * 16);
  us8 c1 = *reinterpret_cast<const us8*>(rowk0 + half * 16 + 8);
  us16 t = __builtin_shufflevector(c0, c1, 0,1,2,3,4,5,6,7,8,9,10,11,12,13,14,15);
  return __builtin_bit_cast(bf16x16, t);
}

// B-fragment (32x16 bf16, KxN): lane l holds column N=l&15, rows K=half*16+i.
// `p` must already point at W_row(N) + k0 + half*16 (16 contiguous bf16).
__device__ __forceinline__ bf16x16 load_bfrag(const u16* p) {
  return __builtin_bit_cast(bf16x16, *reinterpret_cast<const us16*>(p));
}

__device__ __forceinline__ v8f wmma_bf16(bf16x16 a, bf16x16 b, v8f c) {
  return __builtin_amdgcn_wmma_f32_16x16x32_bf16(false, a, false, b,
                                                 (short)0, c, false, false);
}

// ---------------------------------------------------------------------------
// Generic GEMM: C[m][col_off+n] = sum_k A16[m][k] * W16[n][k] + bias[n]
// A16: MxK bf16 row-major, W16: NxK bf16 row-major (B loads contiguous).
// One wave computes a 32(M) x 64(N) tile; K multiple of 32, M multiple of 32.
// ---------------------------------------------------------------------------
__global__ void __launch_bounds__(32)
gemm_bf16_kernel(const u16* __restrict__ A, const u16* __restrict__ W,
                 const float* __restrict__ bias, float* __restrict__ C,
                 int N, int K, int ldc, int col_off) {
  const int lane = threadIdx.x;
  const int r = lane & 15, half = lane >> 4;
  const int n0 = blockIdx.x * 64;
  const int m0 = blockIdx.y * 32;
  v8f z = {0.f,0.f,0.f,0.f,0.f,0.f,0.f,0.f};
  v8f acc0[4] = {z, z, z, z};
  v8f acc1[4] = {z, z, z, z};
  const u16* arow0 = A + (size_t)(m0 + r) * K;
  const u16* arow1 = A + (size_t)(m0 + 16 + r) * K;
  for (int k0 = 0; k0 < K; k0 += 32) {
    bf16x16 a0 = load_afrag(arow0 + k0, half);
    bf16x16 a1 = load_afrag(arow1 + k0, half);
#pragma unroll
    for (int j = 0; j < 4; ++j) {
      bf16x16 b = load_bfrag(W + (size_t)(n0 + j * 16 + r) * K + k0 + half * 16);
      acc0[j] = wmma_bf16(a0, b, acc0[j]);
      acc1[j] = wmma_bf16(a1, b, acc1[j]);
    }
  }
#pragma unroll
  for (int j = 0; j < 4; ++j) {
    int nn = n0 + j * 16 + r;
    if (nn < N) {
      float bv = bias[nn];
#pragma unroll
      for (int v = 0; v < 8; ++v) {
        int mm = m0 + v + 8 * half;
        C[(size_t)mm * ldc + col_off + nn]        = acc0[j][v] + bv;
        C[(size_t)(mm + 16) * ldc + col_off + nn] = acc1[j][v] + bv;
      }
    }
  }
}

// ---------------------------------------------------------------------------
// f32 -> bf16 converters
// ---------------------------------------------------------------------------
__global__ void cvt_kernel(const float* __restrict__ in, u16* __restrict__ out, int n) {
  int i = blockIdx.x * 256 + threadIdx.x;
  if (i < n) out[i] = f2bf(in[i]);
}
__global__ void cvtpad_kernel(const float* __restrict__ in, u16* __restrict__ out,
                              int rows, int cols, int R, int Cc) {
  int i = blockIdx.x * 256 + threadIdx.x;
  if (i >= R * Cc) return;
  int rr = i / Cc, cc = i - rr * Cc;
  out[i] = (rr < rows && cc < cols) ? f2bf(in[rr * cols + cc]) : (u16)0;
}
__global__ void zero16_kernel(u16* p, int n) {
  int i = blockIdx.x * 256 + threadIdx.x;
  if (i < n) p[i] = 0;
}

// ---------------------------------------------------------------------------
// EW1: per token. RMS-norm the 512-dim latent, build kvn=[t, sp, 0-pad] bf16,
// and rope k_pe (64 dims) -> kpe_rot f32.  One wave per token.
// ---------------------------------------------------------------------------
__global__ void __launch_bounds__(32)
ew1_kernel(const float* __restrict__ kvfull, const float* __restrict__ kv_norm_w,
           const float* __restrict__ fcos, const float* __restrict__ fsin,
           u16* __restrict__ kvn16, float* __restrict__ kpe_rot) {
  int tok = blockIdx.x;            // 0..4095
  int lane = threadIdx.x;
  int s = tok & 1023;
  const float* row = kvfull + (size_t)tok * 576;
  float kvv[16]; float loc = 0.f;
#pragma unroll
  for (int i = 0; i < 16; ++i) { kvv[i] = row[lane * 16 + i]; loc += kvv[i] * kvv[i]; }
  for (int m = 16; m; m >>= 1) loc += __shfl_xor(loc, m, 32);
  float inv = rsqrtf(loc * (1.0f / 512.0f) + 1e-6f);
  float sp[16]; float ss2 = 0.f;
#pragma unroll
  for (int i = 0; i < 16; ++i) {
    float v = kvv[i] * inv * kv_norm_w[lane * 16 + i];
    sp[i] = v; ss2 += v * v;
  }
  for (int m = 16; m; m >>= 1) ss2 += __shfl_xor(ss2, m, 32);
  float t = sqrtf(ss2 + 1.0f);
  u16* out = kvn16 + (size_t)tok * 544;
  if (lane == 0) out[0] = f2bf(t);
#pragma unroll
  for (int i = 0; i < 16; ++i) out[1 + lane * 16 + i] = f2bf(sp[i]);
  if (lane < 31) out[513 + lane] = 0;
  // rope k_pe: one (x1,x2) pair per lane
  float x1 = row[512 + 2 * lane], x2 = row[512 + 2 * lane + 1];
  float c = fcos[s * 32 + lane], sn = fsin[s * 32 + lane];
  kpe_rot[(size_t)tok * 64 + 2 * lane]     = x1 * c  - x2 * sn;
  kpe_rot[(size_t)tok * 64 + 2 * lane + 1] = x1 * sn + x2 * c;
}

// ---------------------------------------------------------------------------
// EW_Q: per (token,head). rope q_pe, q0=sqrt(|q|^2+1); write q_ext (b,h,s,224)
// layout: [0..191]=q, [192]=q0, [193..223]=0.
// ---------------------------------------------------------------------------
__global__ void __launch_bounds__(32)
ewq_kernel(const float* __restrict__ qraw, const float* __restrict__ fcos,
           const float* __restrict__ fsin, u16* __restrict__ q_ext) {
  int id = blockIdx.x;             // tok*16 + h
  int tok = id >> 4, h = id & 15;
  int b = tok >> 10, s = tok & 1023;
  int lane = threadIdx.x;
  const float* row = qraw + (size_t)tok * 3072 + h * 192;
  u16* out = q_ext + ((size_t)(b * 16 + h) * 1024 + s) * 224;
  float loc = 0.f;
#pragma unroll
  for (int i = 0; i < 4; ++i) {
    float v = row[lane * 4 + i];
    out[lane * 4 + i] = f2bf(v);
    loc += v * v;
  }
  float x1 = row[128 + 2 * lane], x2 = row[128 + 2 * lane + 1];
  float c = fcos[s * 32 + lane], sn = fsin[s * 32 + lane];
  float o1 = x1 * c - x2 * sn, o2 = x1 * sn + x2 * c;
  out[128 + 2 * lane] = f2bf(o1);
  out[128 + 2 * lane + 1] = f2bf(o2);
  loc += o1 * o1 + o2 * o2;
  for (int m = 16; m; m >>= 1) loc += __shfl_xor(loc, m, 32);
  float q0 = sqrtf(loc + 1.0f);
  if (lane == 0) out[192] = f2bf(q0);
  if (lane < 31) out[193 + lane] = 0;
}

// ---------------------------------------------------------------------------
// EW_K: per (token,head). Assemble k_ext=[k_nope, k_pe_rot, -k0, 0-pad] and
// V^T extended: v_extT[b,h][j][s], j=0..127 spatial, 128=v0, 129..143=0.
// ---------------------------------------------------------------------------
__global__ void __launch_bounds__(32)
ewk_kernel(const float* __restrict__ kv2, const float* __restrict__ kpe_rot,
           u16* __restrict__ k_ext, u16* __restrict__ v_extT) {
  int id = blockIdx.x;
  int tok = id >> 4, h = id & 15;
  int b = tok >> 10, s = tok & 1023;
  int lane = threadIdx.x;
  const float* row = kv2 + (size_t)tok * 4096 + h * 256;
  u16* kout = k_ext + ((size_t)(b * 16 + h) * 1024 + s) * 224;
  float loc = 0.f;
#pragma unroll
  for (int i = 0; i < 4; ++i) {
    float v = row[lane * 4 + i];
    kout[lane * 4 + i] = f2bf(v);
    loc += v * v;
  }
#pragma unroll
  for (int i = 0; i < 2; ++i) {
    float v = kpe_rot[(size_t)tok * 64 + lane * 2 + i];
    kout[128 + lane * 2 + i] = f2bf(v);
    loc += v * v;
  }
  for (int m = 16; m; m >>= 1) loc += __shfl_xor(loc, m, 32);
  float k0 = sqrtf(loc + 1.0f);
  if (lane == 0) kout[192] = f2bf(-k0);       // Minkowski sign folded here
  if (lane < 31) kout[193 + lane] = 0;

  u16* vout = v_extT + (size_t)(b * 16 + h) * 144 * 1024;
  float vloc = 0.f;
#pragma unroll
  for (int i = 0; i < 4; ++i) {
    float v = row[128 + lane * 4 + i];
    vout[(size_t)(lane * 4 + i) * 1024 + s] = f2bf(v);
    vloc += v * v;
  }
  for (int m = 16; m; m >>= 1) vloc += __shfl_xor(vloc, m, 32);
  float v0 = sqrtf(vloc + 1.0f);
  if (lane == 0) vout[(size_t)128 * 1024 + s] = f2bf(v0);
  if (lane >= 1 && lane < 16) vout[(size_t)(128 + lane) * 1024 + s] = 0;
}

// ---------------------------------------------------------------------------
// Flash attention, one wave per (b,h, 16-row q-tile). K blocks of 32 keys.
// QK on WMMA (K=224 incl. Minkowski time dim), online softmax in f32,
// P transposed through 1KB LDS, PV on WMMA (Vt pre-transposed, N=144).
// Epilogue: Lorentz centroid normalization, write cent bf16 (tok x 2080).
// ---------------------------------------------------------------------------
__global__ void __launch_bounds__(32)
attn_kernel(const u16* __restrict__ q_ext, const u16* __restrict__ k_ext,
            const u16* __restrict__ v_extT, u16* __restrict__ cent16) {
  const int qtile = blockIdx.x;    // 0..63
  const int bh = blockIdx.y;       // 0..63
  const int h = bh & 15, b = bh >> 4;
  const int lane = threadIdx.x, r = lane & 15, half = lane >> 4;
  const int qbase = qtile * 16;
  __shared__ u16 plds[16 * 32];

  const u16* Q  = q_ext  + (size_t)bh * 1024 * 224;
  const u16* Kx = k_ext  + (size_t)bh * 1024 * 224;
  const u16* Vt = v_extT + (size_t)bh * 144 * 1024;

  bf16x16 qfrag[7];
#pragma unroll
  for (int kt = 0; kt < 7; ++kt)
    qfrag[kt] = load_afrag(Q + (size_t)(qbase + r) * 224 + kt * 32, half);

  v8f z = {0.f,0.f,0.f,0.f,0.f,0.f,0.f,0.f};
  v8f accO[9];
#pragma unroll
  for (int jt = 0; jt < 9; ++jt) accO[jt] = z;
  float mrun[8], lrun[8];
#pragma unroll
  for (int v = 0; v < 8; ++v) { mrun[v] = -1e30f; lrun[v] = 0.f; }

  const int nkb = qbase / 32 + 1;           // causal: blocks covering keys<=q
  for (int kb = 0; kb < nkb; ++kb) {
    const int kbase = kb * 32;
    v8f s0 = z, s1 = z;
#pragma unroll
    for (int kt = 0; kt < 7; ++kt) {
      bf16x16 b0 = load_bfrag(Kx + (size_t)(kbase + r) * 224 + kt * 32 + half * 16);
      s0 = wmma_bf16(qfrag[kt], b0, s0);
      bf16x16 b1 = load_bfrag(Kx + (size_t)(kbase + 16 + r) * 224 + kt * 32 + half * 16);
      s1 = wmma_bf16(qfrag[kt], b1, s1);
    }
    const int key0 = kbase + r, key1 = kbase + 16 + r;
#pragma unroll
    for (int v = 0; v < 8; ++v) {
      int qrow = qbase + v + 8 * half;
      float a  = 2.0f + 2.0f * SCALE_QK * s0[v] + ((key0 > qrow) ? -1e9f : 0.f);
      float bb = 2.0f + 2.0f * SCALE_QK * s1[v] + ((key1 > qrow) ? -1e9f : 0.f);
      float mx = fmaxf(a, bb);
      for (int m = 8; m; m >>= 1) mx = fmaxf(mx, __shfl_xor(mx, m, 32));
      float newm  = fmaxf(mrun[v], mx);
      float alpha = __expf(mrun[v] - newm);
      float p0 = __expf(a - newm), p1 = __expf(bb - newm);
      float rs = p0 + p1;
      for (int m = 8; m; m >>= 1) rs += __shfl_xor(rs, m, 32);
      lrun[v] = lrun[v] * alpha + rs;
      mrun[v] = newm;
#pragma unroll
      for (int jt = 0; jt < 9; ++jt) accO[jt][v] *= alpha;
      plds[(v + 8 * half) * 32 + r]      = f2bf(p0);   // C-layout -> row major
      plds[(v + 8 * half) * 32 + 16 + r] = f2bf(p1);
    }
    __syncthreads();
    bf16x16 pfrag = load_afrag(&plds[r * 32], half);   // row-major -> A layout
#pragma unroll
    for (int jt = 0; jt < 9; ++jt) {
      bf16x16 vb = load_bfrag(Vt + (size_t)(jt * 16 + r) * 1024 + kbase + half * 16);
      accO[jt] = wmma_bf16(pfrag, vb, accO[jt]);
    }
    __syncthreads();
  }

  // epilogue: avg = O/l, Lorentz centroid normalize, store cent (bf16)
#pragma unroll
  for (int v = 0; v < 8; ++v) {
    float linv = 1.0f / lrun[v];
    float avg[9];
#pragma unroll
    for (int jt = 0; jt < 9; ++jt) avg[jt] = accO[jt][v] * linv;
    float t = (r == 0) ? avg[8] : 0.f;                 // time comp (j=128)
    for (int m = 8; m; m >>= 1) t += __shfl_xor(t, m, 32);
    float ls = 0.f;
#pragma unroll
    for (int jt = 0; jt < 8; ++jt) ls += avg[jt] * avg[jt];
    for (int m = 8; m; m >>= 1) ls += __shfl_xor(ls, m, 32);
    float negl2 = t * t - ls;
    float rinv = rsqrtf(fmaxf(fabsf(negl2), 1e-8f));   // C==1 -> sqrt(C)=1
    int tok = b * 1024 + qbase + v + 8 * half;
    u16* crow = cent16 + (size_t)tok * 2080 + h * 129;
#pragma unroll
    for (int jt = 0; jt < 8; ++jt)
      crow[1 + jt * 16 + r] = f2bf(avg[jt] * rinv);
    if (r == 0) crow[0] = f2bf(t * rinv);
  }
}

// ---------------------------------------------------------------------------
// Final: out[tok][0] = sqrt( sum_{j=1..2047} out[tok][j]^2 + 1 )
// ---------------------------------------------------------------------------
__global__ void __launch_bounds__(256)
rownorm_kernel(float* __restrict__ out) {
  int tok = blockIdx.x, tid = threadIdx.x;
  __shared__ float red[256];
  float* row = out + (size_t)tok * 2048;
  float s = 0.f;
  for (int i = 1 + tid; i < 2048; i += 256) { float v = row[i]; s += v * v; }
  red[tid] = s; __syncthreads();
  for (int k = 128; k; k >>= 1) { if (tid < k) red[tid] += red[tid + k]; __syncthreads(); }
  if (tid == 0) row[0] = sqrtf(red[0] + 1.0f);
}

// ---------------------------------------------------------------------------
// Host orchestration
// ---------------------------------------------------------------------------
extern "C" void kernel_launch(void* const* d_in, const int* in_sizes, int n_in,
                              void* d_out, int out_size, void* d_ws, size_t ws_size,
                              hipStream_t stream) {
  (void)in_sizes; (void)n_in; (void)out_size; (void)ws_size;
  const float* x        = (const float*)d_in[0];
  const float* fcos     = (const float*)d_in[1];
  const float* fsin     = (const float*)d_in[2];
  /* d_in[3] = mask: causal mask implemented analytically in attn_kernel */
  const float* wq_w     = (const float*)d_in[4];
  const float* wq_b     = (const float*)d_in[5];
  const float* wkva_w   = (const float*)d_in[6];
  const float* wkva_b   = (const float*)d_in[7];
  const float* kvnorm_w = (const float*)d_in[8];
  const float* wkvb_w   = (const float*)d_in[9];
  const float* wkvb_b   = (const float*)d_in[10];
  const float* wo_w     = (const float*)d_in[11];
  const float* wo_b     = (const float*)d_in[12];
  float* out = (float*)d_out;

  char* base = (char*)d_ws;
  size_t off = 0;
  auto alloc = [&](size_t bytes) {
    void* p = base + off;
    off += (bytes + 255) & ~(size_t)255;
    return p;
  };
  u16*   xb      = (u16*)  alloc((size_t)4096 * 2048 * 2);   // x bf16
  u16*   wq16    = (u16*)  alloc((size_t)3072 * 2048 * 2);
  u16*   wkva16  = (u16*)  alloc((size_t)576  * 2048 * 2);
  u16*   wkvb16  = (u16*)  alloc((size_t)4096 * 544  * 2);   // K padded 513->544
  u16*   wo16    = (u16*)  alloc((size_t)2048 * 2080 * 2);   // padded 2047x2064
  float* qraw    = (float*)alloc((size_t)4096 * 3072 * 4);
  float* kvfull  = (float*)alloc((size_t)4096 * 576  * 4);
  u16*   kvn16   = (u16*)  alloc((size_t)4096 * 544  * 2);
  float* kperot  = (float*)alloc((size_t)4096 * 64   * 4);
  float* kv2     = (float*)alloc((size_t)4096 * 4096 * 4);
  u16*   qext    = (u16*)  alloc((size_t)64 * 1024 * 224 * 2);
  u16*   kext    = (u16*)  alloc((size_t)64 * 1024 * 224 * 2);
  u16*   vextT   = (u16*)  alloc((size_t)64 * 144  * 1024 * 2);
  u16*   cent16  = (u16*)  alloc((size_t)4096 * 2080 * 2);

  auto blocks = [](size_t n) { return (unsigned)((n + 255) / 256); };

  // weight / activation conversion (bf16)
  cvt_kernel<<<blocks((size_t)4096*2048), 256, 0, stream>>>(x,      xb,     4096*2048);
  cvt_kernel<<<blocks((size_t)3072*2048), 256, 0, stream>>>(wq_w,   wq16,   3072*2048);
  cvt_kernel<<<blocks((size_t)576*2048),  256, 0, stream>>>(wkva_w, wkva16, 576*2048);
  cvtpad_kernel<<<blocks((size_t)4096*544), 256, 0, stream>>>(wkvb_w, wkvb16, 4096, 513, 4096, 544);
  cvtpad_kernel<<<blocks((size_t)2048*2080),256, 0, stream>>>(wo_w,   wo16,   2047, 2064, 2048, 2080);
  zero16_kernel<<<blocks((size_t)4096*2080),256, 0, stream>>>(cent16, 4096*2080);

  // GEMM1: q = x @ wq^T          (4096 x 3072, K=2048)
  gemm_bf16_kernel<<<dim3(48, 128), 32, 0, stream>>>(xb, wq16, wq_b, qraw, 3072, 2048, 3072, 0);
  // GEMM2: kv_full = x @ wkva^T  (4096 x 576, K=2048)
  gemm_bf16_kernel<<<dim3(9, 128), 32, 0, stream>>>(xb, wkva16, wkva_b, kvfull, 576, 2048, 576, 0);
  // RMS-norm latent + project + rope(k_pe)
  ew1_kernel<<<4096, 32, 0, stream>>>(kvfull, kvnorm_w, fcos, fsin, kvn16, kperot);
  // GEMM3: kv2 = kvn @ wkvb^T    (4096 x 4096, K=544 padded)
  gemm_bf16_kernel<<<dim3(64, 128), 32, 0, stream>>>(kvn16, wkvb16, wkvb_b, kv2, 4096, 544, 4096, 0);
  // rope(q_pe) + project -> q_ext ;  assemble k_ext / v_extT
  ewq_kernel<<<65536, 32, 0, stream>>>(qraw, fcos, fsin, qext);
  ewk_kernel<<<65536, 32, 0, stream>>>(kv2, kperot, kext, vextT);
  // flash attention + Lorentz centroid -> cent16
  attn_kernel<<<dim3(64, 64), 32, 0, stream>>>(qext, kext, vextT, cent16);
  // GEMM4: y = cent @ wo^T -> out[:,1:2048]  (N=2047, K=2080 padded)
  gemm_bf16_kernel<<<dim3(32, 128), 32, 0, stream>>>(cent16, wo16, wo_b, out, 2047, 2080, 2048, 1);
  // out[:,0] = sqrt(|y|^2 + 1)
  rownorm_kernel<<<4096, 256, 0, stream>>>(out);
}
